// CasSamplingNet_45784351375505
// MI455X (gfx1250) — matrix operand order repaired and checked by reading, not compiled, not run on hardware
//
#include <hip/hip_runtime.h>
#include <hip/hip_bf16.h>
#include <math.h>

// ---------------------------------------------------------------------------
// CasSamplingNet fused implementation for MI455X (gfx1250, wave32, WMMA).
//
// Exploits rank-1 structure of GAT layer 1 (in_features==1) and rank-2
// structure of GAT layer 2 to avoid the naive 8.6 GFLOP / 134 MB Wh path.
// GEMM-shaped stages (attention pooling + FL MLP) use v_wmma_f32_16x16x32_f16
// with B pre-transposed to fp16 column-major so every fragment load is a
// pair of global_load_b128 (no scalar u16 gather).
// ---------------------------------------------------------------------------

typedef _Float16 h16;
typedef __attribute__((ext_vector_type(16))) _Float16 v16h;
typedef __attribute__((ext_vector_type(8)))  _Float16 v8h;
typedef __attribute__((ext_vector_type(8)))  float    v8f;

#define NEGV (-9e15f)

__device__ __forceinline__ float sigm(float x) { return 1.0f / (1.0f + __expf(-x)); }

// ---------------------------------------------------------------------------
// K0: weight-derived constants
//   cst[0]=c1  cst[1]=c2  cst[2]=pa1 cst[3]=na1 cst[4]=pa2 cst[5]=na2
//   cst[6..13]=P[g]  cst[14..21]=N[g]
// ---------------------------------------------------------------------------
__global__ void k_consts(const float* __restrict__ W1, const float* __restrict__ a1,
                         const float* __restrict__ W2, const float* __restrict__ a2,
                         float* __restrict__ cst)
{
    __shared__ float red[256];
    __shared__ float red8[256 * 8];
    __shared__ float Ps[8], Ns[8];
    const int t = threadIdx.x;

    float pc1 = 0.f, pc2 = 0.f;
    float p8[8] = {0,0,0,0,0,0,0,0};
    float n8[8] = {0,0,0,0,0,0,0,0};
    for (int f = t; f < 512; f += 256) {
        float w = W1[f];
        pc1 += w * a1[f];
        pc2 += w * a1[512 + f];
        float wp = fmaxf(w, 0.f), wn = fminf(w, 0.f);
        #pragma unroll
        for (int g = 0; g < 8; ++g) {
            float w2 = W2[f * 8 + g];
            p8[g] += wp * w2;
            n8[g] += wn * w2;
        }
    }
    // deterministic reductions
    #pragma unroll
    for (int g = 0; g < 8; ++g) red8[t * 8 + g] = p8[g];
    __syncthreads();
    if (t < 8) { float s = 0.f; for (int r = 0; r < 256; ++r) s += red8[r * 8 + t]; Ps[t] = s; }
    __syncthreads();
    #pragma unroll
    for (int g = 0; g < 8; ++g) red8[t * 8 + g] = n8[g];
    __syncthreads();
    if (t < 8) { float s = 0.f; for (int r = 0; r < 256; ++r) s += red8[r * 8 + t]; Ns[t] = s; }

    red[t] = pc1; __syncthreads();
    for (int st = 128; st > 0; st >>= 1) { if (t < st) red[t] += red[t + st]; __syncthreads(); }
    if (t == 0) cst[0] = red[0];
    __syncthreads();
    red[t] = pc2; __syncthreads();
    for (int st = 128; st > 0; st >>= 1) { if (t < st) red[t] += red[t + st]; __syncthreads(); }
    if (t == 0) cst[1] = red[0];
    __syncthreads();
    if (t == 0) {
        float pa1 = 0.f, na1 = 0.f, pa2 = 0.f, na2 = 0.f;
        for (int g = 0; g < 8; ++g) {
            cst[6 + g]  = Ps[g];
            cst[14 + g] = Ns[g];
            pa1 += Ps[g] * a2[g];     na1 += Ns[g] * a2[g];
            pa2 += Ps[g] * a2[8 + g]; na2 += Ns[g] * a2[8 + g];
        }
        cst[2] = pa1; cst[3] = na1; cst[4] = pa2; cst[5] = na2;
    }
}

// ---------------------------------------------------------------------------
// K1: fused double-GAT, one workgroup (128 threads) per batch element.
// LDS: x[128] s[128] wp[128] wn[128] att[128*129]  (~68 KB, pad 129 so that
// column writes (thread=j) and row reads (thread=i) are bank-conflict free).
// Adjacency is read exactly once; mask kept in 4 VGPR bitwords per thread.
// ---------------------------------------------------------------------------
__global__ void k_gat(const int* __restrict__ adj, const float* __restrict__ feat,
                      const float* __restrict__ cst, h16* __restrict__ hmean16)
{
    extern __shared__ float sm[];
    float* xsh  = sm;           // 128
    float* s_sh = sm + 128;     // 128
    float* wpv  = sm + 256;     // 128
    float* wnv  = sm + 384;     // 128
    float* attm = sm + 512;     // 128*129

    const int j = threadIdx.x;
    const int b = blockIdx.x;
    const float c1 = cst[0], c2 = cst[1];
    const float pa1 = cst[2], na1 = cst[3], pa2 = cst[4], na2 = cst[5];

    xsh[j] = feat[b * 128 + j];
    __syncthreads();
    const float xj = xsh[j];

    // ---- layer 1: column (dim=1) softmax of leaky(c1*x_i + c2*x_j) ----
    unsigned mk0 = 0, mk1 = 0, mk2 = 0, mk3 = 0;
    const int* arow = adj + (size_t)b * 128 * 128 + j;
    const float e2 = c2 * xj;
    float m = NEGV;
    for (int i = 0; i < 128; ++i) {
        int a = arow[i * 128];          // coalesced across j
        float e = NEGV;
        if (a > 0) {
            if (i < 32) mk0 |= (1u << i);
            else if (i < 64) mk1 |= (1u << (i - 32));
            else if (i < 96) mk2 |= (1u << (i - 64));
            else mk3 |= (1u << (i - 96));
            float tt = c1 * xsh[i] + e2;
            e = tt >= 0.f ? tt : 0.5f * tt;
        }
        m = fmaxf(m, e);
    }
    float sum = 0.f;
    for (int i = 0; i < 128; ++i) {
        unsigned w = (i < 32) ? mk0 : (i < 64) ? mk1 : (i < 96) ? mk2 : mk3;
        float e = NEGV;
        if ((w >> (i & 31)) & 1u) {
            float tt = c1 * xsh[i] + e2;
            e = tt >= 0.f ? tt : 0.5f * tt;
        }
        float p = __expf(e - m);
        sum += p;
        attm[i * 129 + j] = p;          // unnormalized; fold 1/sum into wpv
    }
    wpv[j] = xj / sum;
    __syncthreads();

    // ---- row pass: s_i = sum_j att1[i,j] * x_j  (thread index acts as row i)
    float s = 0.f;
    for (int jj = 0; jj < 128; ++jj) s += attm[j * 129 + jj] * wpv[jj];
    s_sh[j] = s;
    __syncthreads();

    // ---- layer 2: same mask, scores from pa/na scalars ----
    const float sj = s;
    const float e2p = sj >= 0.f ? pa2 * sj : na2 * sj;
    m = NEGV;
    for (int i = 0; i < 128; ++i) {
        unsigned w = (i < 32) ? mk0 : (i < 64) ? mk1 : (i < 96) ? mk2 : mk3;
        float e = NEGV;
        if ((w >> (i & 31)) & 1u) {
            float si = s_sh[i];
            float e1p = si >= 0.f ? pa1 * si : na1 * si;
            float tt = e1p + e2p;
            e = tt >= 0.f ? tt : 0.5f * tt;
        }
        m = fmaxf(m, e);
    }
    sum = 0.f;
    for (int i = 0; i < 128; ++i) {
        unsigned w = (i < 32) ? mk0 : (i < 64) ? mk1 : (i < 96) ? mk2 : mk3;
        float e = NEGV;
        if ((w >> (i & 31)) & 1u) {
            float si = s_sh[i];
            float e1p = si >= 0.f ? pa1 * si : na1 * si;
            float tt = e1p + e2p;
            e = tt >= 0.f ? tt : 0.5f * tt;
        }
        float p = __expf(e - m);
        sum += p;
        attm[i * 129 + j] = p;
    }
    float inv = 1.f / sum;
    wpv[j] = fmaxf(sj, 0.f) * inv;
    wnv[j] = fminf(sj, 0.f) * inv;
    __syncthreads();

    // ---- row pass 2 + mean over g of relu(P_g*u + N_g*v) ----
    float u = 0.f, v = 0.f;
    for (int jj = 0; jj < 128; ++jj) {
        float p = attm[j * 129 + jj];
        u += p * wpv[jj];
        v += p * wnv[jj];
    }
    float hm = 0.f;
    #pragma unroll
    for (int g = 0; g < 8; ++g)
        hm += fmaxf(cst[6 + g] * u + cst[14 + g] * v, 0.f);
    hm *= 0.125f;
    hmean16[b * 128 + j] = (h16)hm;
}

// ---------------------------------------------------------------------------
// fp32 -> fp16 conversion (row-major copy, for A operands)
// ---------------------------------------------------------------------------
__global__ void k_cvt16(const float* __restrict__ src, h16* __restrict__ dst, int n)
{
    int i = blockIdx.x * blockDim.x + threadIdx.x;
    if (i < n) dst[i] = (h16)src[i];
}

// ---------------------------------------------------------------------------
// fp32 [K,N] -> fp16 [N,K] transposed conversion (for B operands): makes the
// per-lane WMMA B-fragment (two runs of 8 consecutive K at fixed N) two contiguous 16 B pieces.
// ---------------------------------------------------------------------------
__global__ void k_cvt16t(const float* __restrict__ src, h16* __restrict__ dst,
                         int K, int N)
{
    int i = blockIdx.x * blockDim.x + threadIdx.x;
    if (i < K * N) {
        int k = i / N, n = i - k * N;       // coalesced read, strided write
        dst[n * K + k] = (h16)src[i];
    }
}

// ---------------------------------------------------------------------------
// Generic WMMA GEMM: C[M,N] = A[M,K] @ Bt[N,K]^T, f16 in / f32 accumulate.
// Block = 128 threads = 4 waves; each wave owns one 16x16 C tile.
// Grid = (M/64, N/16).  mode: 0=none, 1=bias+relu, 2=bias+tanh.
// Fragment layouts per CDNA5 ISA:
//   A 16x32 f16 : lanes<16 K=[0..7]+[16..23], lanes>=16 K=[8..15]+[24..31]
//   B 32x16 f16 : lanes<16 K=[0..7]+[16..23], lanes>=16 K=[8..15]+[24..31] (N = lane%16)
//   C 16x16 f32 : VGPR v -> row v + 8*(lane>=16), col lane%16
// Both fragments load as two 16-byte vectors (global_load_b128).
// ---------------------------------------------------------------------------
__global__ void k_gemm_wmma(const h16* __restrict__ A, const h16* __restrict__ Bt,
                            float* __restrict__ Cf, h16* __restrict__ Ch,
                            const float* __restrict__ bias,
                            int M, int N, int K, int mode)
{
    const int lane = threadIdx.x & 31;
    const int wave = threadIdx.x >> 5;
    const int m0 = blockIdx.x * 64 + wave * 16;
    const int n0 = blockIdx.y * 16;
    if (m0 >= M) return;                 // wave-uniform, EXEC stays all-ones
    const int hi = lane >> 4;            // lane half (0/1)
    const int ln = lane & 15;

    const h16* Ap = A  + (size_t)(m0 + ln) * K + hi * 8;   // + k0 per step
    const h16* Bp = Bt + (size_t)(n0 + ln) * K + hi * 8;   // + k0 per step

    v8f acc = {};
    for (int k0 = 0; k0 < K; k0 += 32) {
        v8h alo = *(const v8h*)(Ap + k0);        // K = base .. base+7
        v8h ahi = *(const v8h*)(Ap + k0 + 16);   // K = base+16 .. base+23
        v8h blo = *(const v8h*)(Bp + k0);        // K = base .. base+7
        v8h bhi = *(const v8h*)(Bp + k0 + 16);   // K = base+16 .. base+23
        v16h a = __builtin_shufflevector(alo, ahi, 0,1,2,3,4,5,6,7,
                                                   8,9,10,11,12,13,14,15);
        v16h bfr = __builtin_shufflevector(blo, bhi, 0,1,2,3,4,5,6,7,
                                                     8,9,10,11,12,13,14,15);
        acc = __builtin_amdgcn_wmma_f32_16x16x32_f16(
            false, a, false, bfr, (short)0, acc, false, false);
    }
    #pragma unroll
    for (int v = 0; v < 8; ++v) {
        int mr = m0 + v + hi * 8;
        int nc = n0 + ln;
        float val = acc[v];
        if (bias) val += bias[nc];
        if (mode == 1) val = fmaxf(val, 0.f);
        else if (mode == 2) val = tanhf(val);
        if (Cf) Cf[(size_t)mr * N + nc] = val;
        if (Ch) Ch[(size_t)mr * N + nc] = (h16)val;
    }
}

// ---------------------------------------------------------------------------
// K2: fused 2-layer LSTM, hidden=1, one thread per batch. Writes hcat[:,128+t].
// ---------------------------------------------------------------------------
__global__ void k_lstm(const float* __restrict__ x,
                       const float* __restrict__ wih0, const float* __restrict__ whh0,
                       const float* __restrict__ bih0, const float* __restrict__ bhh0,
                       const float* __restrict__ wih1, const float* __restrict__ whh1,
                       const float* __restrict__ bih1, const float* __restrict__ bhh1,
                       float* __restrict__ hcat)
{
    int b = blockIdx.x * blockDim.x + threadIdx.x;
    if (b >= 512) return;
    float wi0[4], wh0[4], bb0[4], wi1[4], wh1[4], bb1[4];
    #pragma unroll
    for (int k = 0; k < 4; ++k) {
        wi0[k] = wih0[k]; wh0[k] = whh0[k]; bb0[k] = bih0[k] + bhh0[k];
        wi1[k] = wih1[k]; wh1[k] = whh1[k]; bb1[k] = bih1[k] + bhh1[k];
    }
    float h0 = 0.f, c0 = 0.f, h1 = 0.f, c1 = 0.f;
    for (int t = 0; t < 64; ++t) {
        float xt = x[b * 64 + t];
        float gi = sigm(xt * wi0[0] + h0 * wh0[0] + bb0[0]);
        float gf = sigm(xt * wi0[1] + h0 * wh0[1] + bb0[1]);
        float gg = tanhf(xt * wi0[2] + h0 * wh0[2] + bb0[2]);
        float go = sigm(xt * wi0[3] + h0 * wh0[3] + bb0[3]);
        c0 = gf * c0 + gi * gg;
        h0 = go * tanhf(c0);
        gi = sigm(h0 * wi1[0] + h1 * wh1[0] + bb1[0]);
        gf = sigm(h0 * wi1[1] + h1 * wh1[1] + bb1[1]);
        gg = tanhf(h0 * wi1[2] + h1 * wh1[2] + bb1[2]);
        go = sigm(h0 * wi1[3] + h1 * wh1[3] + bb1[3]);
        c1 = gf * c1 + gi * gg;
        h1 = go * tanhf(c1);
        hcat[b * 192 + 128 + t] = h1;
    }
}

// ---------------------------------------------------------------------------
// K3: attention-pooling combine (pair softmax + faithful [2,128]->[128,2]
// reshape of the Wv branch). Writes hcat[:, :128].
// ---------------------------------------------------------------------------
__global__ void k_combine(const float* __restrict__ HW, const float* __restrict__ TW,
                          const float* __restrict__ HV, const float* __restrict__ TV,
                          float* __restrict__ hcat)
{
    int idx = blockIdx.x * blockDim.x + threadIdx.x;
    if (idx >= 512 * 128) return;
    int b = idx >> 7, i = idx & 127;
    float a0 = HW[idx], a1 = TW[idx];
    float m  = fmaxf(a0, a1);
    float e0 = __expf(a0 - m), e1 = __expf(a1 - m);
    float inv = 1.f / (e0 + e1);
    int f0 = 2 * i, f1 = 2 * i + 1;
    float hv0 = (f0 < 128) ? HV[b * 128 + f0] : TV[b * 128 + f0 - 128];
    float hv1 = (f1 < 128) ? HV[b * 128 + f1] : TV[b * 128 + f1 - 128];
    hcat[b * 192 + i] = (fmaxf(hv0, 0.f) * e0 + fmaxf(hv1, 0.f) * e1) * inv;
}

// ---------------------------------------------------------------------------
// K4: MHA (embed_dim=1) + residual. a[i,j] = softmax_j(q_i * k_j). One block
// per batch, 192 threads (row-softmax needs no cross-thread reduction).
// Emits fp16 for the MLP GEMM.
// ---------------------------------------------------------------------------
__global__ void k_mha(const float* __restrict__ hcat,
                      const float* __restrict__ inw, const float* __restrict__ inb,
                      const float* __restrict__ outw, const float* __restrict__ outb,
                      h16* __restrict__ g16)
{
    __shared__ float karr[192];
    __shared__ float varr[192];
    const int t = threadIdx.x, b = blockIdx.x;
    float h = hcat[b * 192 + t];
    float q = h * inw[0] + inb[0];
    karr[t] = h * inw[1] + inb[1];
    varr[t] = h * inw[2] + inb[2];
    __syncthreads();
    float m = -3.4e38f;
    for (int j = 0; j < 192; ++j) m = fmaxf(m, q * karr[j]);
    float se = 0.f, sv = 0.f;
    for (int j = 0; j < 192; ++j) {
        float e = __expf(q * karr[j] - m);
        se += e;
        sv += e * varr[j];
    }
    float ho = (sv / se) * outw[0] + outb[0];
    g16[b * 192 + t] = (h16)(ho + h);
}

// ---------------------------------------------------------------------------
// K5: final 32->1 layer, relu.
// ---------------------------------------------------------------------------
__global__ void k_fl3(const float* __restrict__ act2, const float* __restrict__ w,
                      const float* __restrict__ bias, float* __restrict__ out)
{
    int b = blockIdx.x * blockDim.x + threadIdx.x;
    if (b >= 512) return;
    float acc = bias[0];
    #pragma unroll
    for (int k = 0; k < 32; ++k) acc += act2[b * 32 + k] * w[k];
    out[b] = fmaxf(acc, 0.f);
}

// ---------------------------------------------------------------------------
// Workspace layout (bytes), total ~2.2 MB
// ---------------------------------------------------------------------------
#define OFF_CST    0u          //  256 B: 22 consts
#define OFF_HM16   256u        //  512*128 f16 (A, row-major)
#define OFF_T16    131328u     //  512*128 f16 (A, row-major)
#define OFF_WT16   262400u     //  [N=128,K=128] f16 (B, transposed)
#define OFF_WV16   295168u     //  [N=128,K=128] f16 (B, transposed)
#define OFF_F1W16  327936u     //  [N=128,K=192] f16 (B, transposed)
#define OFF_F2W16  377088u     //  [N=32, K=128] f16 (B, transposed)
#define OFF_HW     385280u     //  512*128 f32
#define OFF_TW     647424u
#define OFF_HV     909568u
#define OFF_TV     1171712u
#define OFF_HCAT   1433856u    //  512*192 f32
#define OFF_G16    1827072u    //  512*192 f16 (A, row-major)
#define OFF_A116   2023680u    //  512*128 f16 (A, row-major)
#define OFF_A2F    2154752u    //  512*32  f32

extern "C" void kernel_launch(void* const* d_in, const int* in_sizes, int n_in,
                              void* d_out, int out_size, void* d_ws, size_t ws_size,
                              hipStream_t stream)
{
    const int*   adj      = (const int*)  d_in[0];
    const float* feat     = (const float*)d_in[1];
    const float* timev    = (const float*)d_in[2];
    const float* ipop     = (const float*)d_in[3];
    const float* W1       = (const float*)d_in[4];
    const float* a1       = (const float*)d_in[5];
    const float* W2       = (const float*)d_in[6];
    const float* a2       = (const float*)d_in[7];
    const float* Weight   = (const float*)d_in[8];
    const float* Wv       = (const float*)d_in[9];
    const float* wih0     = (const float*)d_in[10];
    const float* whh0     = (const float*)d_in[11];
    const float* bih0     = (const float*)d_in[12];
    const float* bhh0     = (const float*)d_in[13];
    const float* wih1     = (const float*)d_in[14];
    const float* whh1     = (const float*)d_in[15];
    const float* bih1     = (const float*)d_in[16];
    const float* bhh1     = (const float*)d_in[17];
    const float* mha_in_w = (const float*)d_in[18];
    const float* mha_in_b = (const float*)d_in[19];
    const float* mha_o_w  = (const float*)d_in[20];
    const float* mha_o_b  = (const float*)d_in[21];
    const float* fl1_w    = (const float*)d_in[22];
    const float* fl1_b    = (const float*)d_in[23];
    const float* fl2_w    = (const float*)d_in[24];
    const float* fl2_b    = (const float*)d_in[25];
    const float* fl3_w    = (const float*)d_in[26];
    const float* fl3_b    = (const float*)d_in[27];

    char* ws = (char*)d_ws;
    float* cst   = (float*)(ws + OFF_CST);
    h16* hm16    = (h16*)(ws + OFF_HM16);
    h16* t16     = (h16*)(ws + OFF_T16);
    h16* wt16    = (h16*)(ws + OFF_WT16);
    h16* wv16    = (h16*)(ws + OFF_WV16);
    h16* f1w16   = (h16*)(ws + OFF_F1W16);
    h16* f2w16   = (h16*)(ws + OFF_F2W16);
    float* HW    = (float*)(ws + OFF_HW);
    float* TW    = (float*)(ws + OFF_TW);
    float* HV    = (float*)(ws + OFF_HV);
    float* TV    = (float*)(ws + OFF_TV);
    float* hcat  = (float*)(ws + OFF_HCAT);
    h16* g16     = (h16*)(ws + OFF_G16);
    h16* a116    = (h16*)(ws + OFF_A116);
    float* a2f   = (float*)(ws + OFF_A2F);

    // weight-derived scalars
    k_consts<<<1, 256, 0, stream>>>(W1, a1, W2, a2, cst);

    // fp16 copies: A operands row-major, B operands transposed [N,K]
    k_cvt16<<<(512 * 128 + 255) / 256, 256, 0, stream>>>(timev, t16, 512 * 128);
    k_cvt16t<<<64, 256, 0, stream>>>(Weight, wt16, 128, 128);
    k_cvt16t<<<64, 256, 0, stream>>>(Wv,     wv16, 128, 128);
    k_cvt16t<<<96, 256, 0, stream>>>(fl1_w,  f1w16, 192, 128);
    k_cvt16t<<<16, 256, 0, stream>>>(fl2_w,  f2w16, 128, 32);

    // fused double-GAT: 1 block per batch, 68 KB dynamic LDS
    const size_t gat_shm = (size_t)(512 + 128 * 129) * sizeof(float);
    k_gat<<<512, 128, gat_shm, stream>>>(adj, feat, cst, hm16);

    // pooling GEMMs: [512,128] @ [128,128], WMMA
    dim3 g88(8, 8), g82(8, 2);
    k_gemm_wmma<<<g88, 128, 0, stream>>>(hm16, wt16, HW, nullptr, nullptr, 512, 128, 128, 0);
    k_gemm_wmma<<<g88, 128, 0, stream>>>(t16,  wt16, TW, nullptr, nullptr, 512, 128, 128, 0);
    k_gemm_wmma<<<g88, 128, 0, stream>>>(hm16, wv16, HV, nullptr, nullptr, 512, 128, 128, 0);
    k_gemm_wmma<<<g88, 128, 0, stream>>>(t16,  wv16, TV, nullptr, nullptr, 512, 128, 128, 0);

    // LSTM (independent branch) writes hcat[:,128:192]
    k_lstm<<<2, 256, 0, stream>>>(ipop, wih0, whh0, bih0, bhh0,
                                  wih1, whh1, bih1, bhh1, hcat);

    // pooling combine writes hcat[:,0:128]
    k_combine<<<256, 256, 0, stream>>>(HW, TW, HV, TV, hcat);

    // degenerate MHA + residual -> f16 activations
    k_mha<<<512, 192, 0, stream>>>(hcat, mha_in_w, mha_in_b, mha_o_w, mha_o_b, g16);

    // FL MLP via WMMA: 192->128 (relu), 128->32 (tanh), 32->1 (relu)
    k_gemm_wmma<<<g88, 128, 0, stream>>>(g16,  f1w16, nullptr, a116, fl1_b, 512, 128, 192, 1);
    k_gemm_wmma<<<g82, 128, 0, stream>>>(a116, f2w16, a2f, nullptr, fl2_b, 512, 32, 128, 2);
    k_fl3<<<2, 256, 0, stream>>>(a2f, fl3_w, fl3_b, (float*)d_out);
}
